// GcnEncoderBlock_83193516523963
// MI455X (gfx1250) — matrix operand order repaired and checked by reading, not compiled, not run on hardware
//
#include <hip/hip_runtime.h>
#include <cstdint>
#include <cstddef>

namespace {

constexpr int kB   = 8;
constexpr int kN   = 2048;   // nodes
constexpr int kFin = 128;
constexpr int kHid = 256;

typedef __attribute__((ext_vector_type(16))) __bf16 v16bf;
typedef __attribute__((ext_vector_type(2)))  __bf16 v2bf;
typedef __attribute__((ext_vector_type(8)))  float  v8f;
typedef __attribute__((ext_vector_type(4)))  int    v4i;

union BFrag {
  v16bf v;
  uint4 q[2];
  unsigned int u[8];
};

// bf16 via truncation: single shift (scalar) / single v_perm_b32 (pair).
__device__ __forceinline__ unsigned short f2bf(float x) {
  return (unsigned short)(__builtin_bit_cast(unsigned int, x) >> 16);
}

// Pack two f32 into a bf16x2 dword.
__device__ __forceinline__ unsigned int packbf2(float lo, float hi) {
#if __has_builtin(__builtin_amdgcn_cvt_pk_bf16_f32)
  v2bf p = __builtin_amdgcn_cvt_pk_bf16_f32(lo, hi);
  return __builtin_bit_cast(unsigned int, p);
#else
  // v_perm_b32: result = { hi[31:16], lo[31:16] }  (one VALU op)
  return __builtin_amdgcn_perm(__builtin_bit_cast(unsigned int, hi),
                               __builtin_bit_cast(unsigned int, lo),
                               0x07060302u);
#endif
}

// ---- async global->LDS staging (gfx1250 GLOBAL_LOAD_ASYNC_TO_LDS_B128) -----
#if __has_builtin(__builtin_amdgcn_global_load_async_to_lds_b128)
#define GCN_ASYNC_LDS 1
#else
#define GCN_ASYNC_LDS 0
#endif

__device__ __forceinline__ void stage16(const unsigned short* __restrict__ g,
                                        unsigned short* __restrict__ l) {
#if GCN_ASYNC_LDS
  __builtin_amdgcn_global_load_async_to_lds_b128(
      (v4i*)g,
      (__attribute__((address_space(3))) v4i*)l, 0, 0);
#else
  *(uint4*)l = *(const uint4*)g;   // global_load_b128 + ds_store_b128
#endif
}

__device__ __forceinline__ void wait_async() {
#if GCN_ASYNC_LDS
#if __has_builtin(__builtin_amdgcn_s_wait_asynccnt)
  __builtin_amdgcn_s_wait_asynccnt(0);
#else
  asm volatile("s_wait_asynccnt 0" ::: "memory");
#endif
#endif
}

// ---- fragment loaders ------------------------------------------------------
// A-fragment (16x32 bf16) from an fp32 row-major source.
// p points at element (row = row0+idx, col = k0 + half*8); layout per ISA:
// lane half h holds K runs [8h, 8h+8) and [16+8h, 16+8h+8), packed 2/VGPR.
__device__ __forceinline__ void loadA_f32(BFrag& f, const float* __restrict__ p) {
  float4 x0 = *(const float4*)(p + 0);
  float4 x1 = *(const float4*)(p + 4);
  float4 x2 = *(const float4*)(p + 16);
  float4 x3 = *(const float4*)(p + 20);
  f.u[0] = packbf2(x0.x, x0.y);
  f.u[1] = packbf2(x0.z, x0.w);
  f.u[2] = packbf2(x1.x, x1.y);
  f.u[3] = packbf2(x1.z, x1.w);
  f.u[4] = packbf2(x2.x, x2.y);
  f.u[5] = packbf2(x2.z, x2.w);
  f.u[6] = packbf2(x3.x, x3.y);
  f.u[7] = packbf2(x3.z, x3.w);
}

// A-fragment from bf16 row-major source; p at (row0+idx, k0 + half*8).
__device__ __forceinline__ void loadA_bf16(BFrag& f, const unsigned short* __restrict__ p) {
  f.q[0] = *(const uint4*)(p);        // K = 8h .. 8h+7
  f.q[1] = *(const uint4*)(p + 16);   // K = 16+8h .. 16+8h+7
}

// B-fragment (32x16 bf16) from a transposed bf16 source srcT[n][k];
// p at (n = col0+idx row of srcT, k0 + half*8): lane half h holds K runs [8h, 8h+8) and [16+8h, 16+8h+8).
__device__ __forceinline__ void loadB_T(BFrag& f, const unsigned short* __restrict__ p) {
  f.q[0] = *(const uint4*)(p);
  f.q[1] = *(const uint4*)(p + 16);
}

__device__ __forceinline__ v8f wmma_bf16(const BFrag& a, const BFrag& b, v8f c) {
  return __builtin_amdgcn_wmma_f32_16x16x32_bf16(false, a.v, false, b.v,
                                                 (short)0, c, false, false);
}

// Store one 16x16 f32 accumulator as bf16 into a transposed [n][m] buffer.
// C/D layout: lane idx = N, VGPR r -> M = 8*half + r  =>  8 contiguous m per lane.
__device__ __forceinline__ void storeC_T(unsigned short* __restrict__ dstT, int ldT,
                                         const v8f& acc, int c0, int row0,
                                         int idx, int half) {
  unsigned short* p = dstT + (size_t)(c0 + idx) * ldT + row0 + half * 8;
  uint4 q;
  q.x = packbf2(acc[0], acc[1]);
  q.y = packbf2(acc[2], acc[3]);
  q.z = packbf2(acc[4], acc[5]);
  q.w = packbf2(acc[6], acc[7]);
  *(uint4*)p = q;
}

// ---- kernels ---------------------------------------------------------------

// dst[c][r] = bf16(src[r][c])   (weight transpose + convert)
__global__ __launch_bounds__(256) void k_transpose_bf16(const float* __restrict__ src,
                                                        unsigned short* __restrict__ dst,
                                                        int rows, int cols) {
  int i = blockIdx.x * blockDim.x + threadIdx.x;
  if (i >= rows * cols) return;
  int r = i / cols, c = i - r * cols;
  dst[(size_t)c * rows + r] = f2bf(src[i]);
}

// Bulk f32 -> bf16 conversion, 8 elements/thread (2x b128 load, 1x b128 store).
__global__ __launch_bounds__(256) void k_conv_bf16(const float* __restrict__ src,
                                                   unsigned short* __restrict__ dst,
                                                   long n8) {
  long i = (long)blockIdx.x * blockDim.x + threadIdx.x;
  if (i >= n8) return;
  const float4* s = (const float4*)src + 2 * i;
  float4 x0 = s[0], x1 = s[1];
  uint4 q;
  q.x = packbf2(x0.x, x0.y);
  q.y = packbf2(x0.z, x0.w);
  q.z = packbf2(x1.x, x1.y);
  q.w = packbf2(x1.z, x1.w);
  ((uint4*)dst)[i] = q;
}

// xwT[b][h][m] = (X[b] @ W1)^T, bf16.  M=2048, N=256, K=128.
__global__ __launch_bounds__(128) void k_xw(const float* __restrict__ x,
                                            const unsigned short* __restrict__ w1T,
                                            unsigned short* __restrict__ xwT) {
  const int lane = threadIdx.x & 31;
  const int wave = threadIdx.x >> 5;
  const int idx  = lane & 15, half = lane >> 4;
  const int b    = blockIdx.z;
  const int row0 = (blockIdx.x * 4 + wave) * 16;
  const int col0 = blockIdx.y * 64;
  const float* pa = x + (size_t)b * kN * kFin + (size_t)(row0 + idx) * kFin + half * 8;
  const unsigned short* pb = w1T + (size_t)(col0 + idx) * kFin + half * 8;
  v8f a0 = {}, a1 = {}, a2 = {}, a3 = {};
#pragma unroll
  for (int k0 = 0; k0 < kFin; k0 += 32) {
    BFrag fa, f0, f1, f2, f3;
    loadA_f32(fa, pa);
    loadB_T(f0, pb);
    loadB_T(f1, pb + 16 * kFin);
    loadB_T(f2, pb + 32 * kFin);
    loadB_T(f3, pb + 48 * kFin);
    a0 = wmma_bf16(fa, f0, a0);
    a1 = wmma_bf16(fa, f1, a1);
    a2 = wmma_bf16(fa, f2, a2);
    a3 = wmma_bf16(fa, f3, a3);
    pa += 32;
    pb += 32;
  }
  unsigned short* dT = xwT + (size_t)b * kHid * kN;
  storeC_T(dT, kN, a0, col0 +  0, row0, idx, half);
  storeC_T(dT, kN, a1, col0 + 16, row0, idx, half);
  storeC_T(dT, kN, a2, col0 + 32, row0, idx, half);
  storeC_T(dT, kN, a3, col0 + 48, row0, idx, half);
}

// hwT[b][h2][m] = (H[b] @ W2)^T, bf16.  M=2048, N=256, K=256, A operand bf16.
__global__ __launch_bounds__(128) void k_hw(const unsigned short* __restrict__ h,
                                            const unsigned short* __restrict__ w2T,
                                            unsigned short* __restrict__ hwT) {
  const int lane = threadIdx.x & 31;
  const int wave = threadIdx.x >> 5;
  const int idx  = lane & 15, half = lane >> 4;
  const int b    = blockIdx.z;
  const int row0 = (blockIdx.x * 4 + wave) * 16;
  const int col0 = blockIdx.y * 64;
  const unsigned short* pa =
      h + (size_t)b * kN * kHid + (size_t)(row0 + idx) * kHid + half * 8;
  const unsigned short* pb = w2T + (size_t)(col0 + idx) * kHid + half * 8;
  v8f a0 = {}, a1 = {}, a2 = {}, a3 = {};
#pragma unroll 4
  for (int k0 = 0; k0 < kHid; k0 += 32) {
    BFrag fa, f0, f1, f2, f3;
    loadA_bf16(fa, pa);
    loadB_T(f0, pb);
    loadB_T(f1, pb + 16 * kHid);
    loadB_T(f2, pb + 32 * kHid);
    loadB_T(f3, pb + 48 * kHid);
    a0 = wmma_bf16(fa, f0, a0);
    a1 = wmma_bf16(fa, f1, a1);
    a2 = wmma_bf16(fa, f2, a2);
    a3 = wmma_bf16(fa, f3, a3);
    pa += 32;
    pb += 32;
  }
  unsigned short* dT = hwT + (size_t)b * kHid * kN;
  storeC_T(dT, kN, a0, col0 +  0, row0, idx, half);
  storeC_T(dT, kN, a1, col0 + 16, row0, idx, half);
  storeC_T(dT, kN, a2, col0 + 32, row0, idx, half);
  storeC_T(dT, kN, a3, col0 + 48, row0, idx, half);
}

// Aggregation: out = relu(A[b] @ srcT^T + bias).  M=2048, N=256, K=2048.
// Block = 128 threads = 4 waves; block tile 128x64; wave tile 32x64
// (2 A-fragments x 4 B-fragments = 8 WMMAs / K-step).
// B tiles (64x32 bf16, 4 KB) are staged once per block through double-buffered
// LDS shared by all 4 waves; staging uses GLOBAL_LOAD_ASYNC_TO_LDS_B128 when
// the toolchain exposes it (ASYNCcnt overlaps next-tile loads with this
// tile's WMMAs; fenced by s_wait_asynccnt just before the barrier).
// ABF16: A operand already bf16 (pure-load hot loop); else fp32 via v_perm.
// OUTF32: f32 row-major (final output); else bf16 row-major (hidden layer H).
template <bool ABF16, bool OUTF32>
__global__ __launch_bounds__(128) void k_agg(const void* __restrict__ adjv,
                                             const unsigned short* __restrict__ srcT,
                                             const float* __restrict__ bias,
                                             void* __restrict__ out) {
  constexpr int kPad = 40;   // LDS row stride in bf16 (80 B: 16 B-aligned, depaired banks)
  __shared__ __align__(16) unsigned short sB[2][64 * kPad];

  const int tid  = threadIdx.x;
  const int lane = tid & 31;
  const int wave = tid >> 5;
  const int idx  = lane & 15, half = lane >> 4;
  const int b    = blockIdx.z;
  const int row0 = (blockIdx.x * 4 + wave) * 32;
  const int col0 = blockIdx.y * 64;

  // --- B staging: 4 KB tile = 256 x 16 B chunks; each thread owns 2 chunks.
  const int sn   = tid >> 2;           // row 0..31 (and +32)
  const int sk   = (tid & 3) * 8;      // 8-element (16 B) column chunk
  const unsigned short* gB0 =
      srcT + (size_t)b * kHid * kN + (size_t)(col0 + sn) * kN + sk;
  const unsigned short* gB1 = gB0 + (size_t)32 * kN;
  const int lB0 = sn * kPad + sk;
  const int lB1 = lB0 + 32 * kPad;

  // --- A pointers (two 16-row fragments per wave)
  const size_t aBase = (size_t)b * kN * kN + (size_t)(row0 + idx) * kN + half * 8;
  const float*          paf = (const float*)adjv + aBase;
  const unsigned short* pah = (const unsigned short*)adjv + aBase;

  v8f acc[2][4] = {};

  // prologue: stage K-step 0
  stage16(gB0, &sB[0][lB0]);
  stage16(gB1, &sB[0][lB1]);
  wait_async();
  __syncthreads();

  int buf = 0;
  for (int k0 = 0; k0 < kN; k0 += 32) {
    if (k0 + 32 < kN) {   // stage next tile into the other buffer
      stage16(gB0 + k0 + 32, &sB[buf ^ 1][lB0]);
      stage16(gB1 + k0 + 32, &sB[buf ^ 1][lB1]);
    }
    BFrag fa0, fa1;
    if (ABF16) {
      loadA_bf16(fa0, pah + k0);
      loadA_bf16(fa1, pah + (size_t)16 * kN + k0);
      __builtin_prefetch(pah + k0 + 64, 0, 0);   // global_prefetch_b8
    } else {
      loadA_f32(fa0, paf + k0);
      loadA_f32(fa1, paf + (size_t)16 * kN + k0);
      __builtin_prefetch(paf + k0 + 32, 0, 0);
    }
#pragma unroll
    for (int j = 0; j < 4; ++j) {
      BFrag fb;
      const unsigned short* lp = &sB[buf][(16 * j + idx) * kPad + half * 8];
      fb.q[0] = *(const uint4*)(lp);      // ds_load_b128
      fb.q[1] = *(const uint4*)(lp + 16); // ds_load_b128
      acc[0][j] = wmma_bf16(fa0, fb, acc[0][j]);
      acc[1][j] = wmma_bf16(fa1, fb, acc[1][j]);
    }
    wait_async();      // next-tile async loads complete (overlapped with WMMAs)
    __syncthreads();
    buf ^= 1;
  }

  // bias + relu + store, row-major [m][n]
#pragma unroll
  for (int i = 0; i < 2; ++i) {
#pragma unroll
    for (int j = 0; j < 4; ++j) {
      const int n  = col0 + 16 * j + idx;
      const int m0 = row0 + 16 * i + half * 8;
      const float bv = bias[n];
#pragma unroll
      for (int r = 0; r < 8; ++r) {
        float vv = fmaxf(acc[i][j][r] + bv, 0.0f);
        if (OUTF32) {
          ((float*)out)[(size_t)b * kN * kHid + (size_t)(m0 + r) * kHid + n] = vv;
        } else {
          ((unsigned short*)out)[(size_t)b * kN * kHid + (size_t)(m0 + r) * kHid + n] =
              f2bf(vv);
        }
      }
    }
  }
}

}  // namespace

extern "C" void kernel_launch(void* const* d_in, const int* in_sizes, int n_in,
                              void* d_out, int out_size, void* d_ws, size_t ws_size,
                              hipStream_t stream) {
  const float* x  = (const float*)d_in[0];   // [8,2048,128]
  const float* a  = (const float*)d_in[1];   // [8,2048,2048]
  const float* w1 = (const float*)d_in[2];   // [128,256]
  const float* b1 = (const float*)d_in[3];   // [256]
  const float* w2 = (const float*)d_in[4];   // [256,256]
  const float* b2 = (const float*)d_in[5];   // [256]

  // workspace layout
  constexpr size_t kOffW1T = 0;                                   //  64 KB
  constexpr size_t kOffW2T = (size_t)(64 << 10);                  // 128 KB
  constexpr size_t kOffXWT = (size_t)(192 << 10);                 //   8 MB
  constexpr size_t kOffH   = kOffXWT + (size_t)kB * kHid * kN * 2;
  constexpr size_t kOffHWT = kOffH   + (size_t)kB * kHid * kN * 2;
  constexpr size_t kOffABF = kOffHWT + (size_t)kB * kHid * kN * 2;
  constexpr size_t kWsFull = kOffABF + (size_t)kB * kN * kN * 2;  // + 64 MB bf16 A

  char* ws = (char*)d_ws;
  unsigned short* w1T = (unsigned short*)(ws + kOffW1T);
  unsigned short* w2T = (unsigned short*)(ws + kOffW2T);
  unsigned short* xwT = (unsigned short*)(ws + kOffXWT);
  unsigned short* H   = (unsigned short*)(ws + kOffH);
  unsigned short* hwT = (unsigned short*)(ws + kOffHWT);
  unsigned short* aBF = (unsigned short*)(ws + kOffABF);
  const bool useABF = (ws_size >= kWsFull);   // ws_size is fixed -> deterministic

  k_transpose_bf16<<<(kFin * kHid + 255) / 256, 256, 0, stream>>>(w1, w1T, kFin, kHid);
  k_transpose_bf16<<<(kHid * kHid + 255) / 256, 256, 0, stream>>>(w2, w2T, kHid, kHid);

  const dim3 gridMM(kN / 64, kHid / 64, kB);   // (32, 4, 8) for the X@W / H@W GEMMs
  const dim3 gridAG(kN / 128, kHid / 64, kB);  // (16, 4, 8) for the aggregations
  k_xw<<<gridMM, 128, 0, stream>>>(x, w1T, xwT);

  if (useABF) {
    const long n8 = (long)kB * kN * kN / 8;
    k_conv_bf16<<<(unsigned)((n8 + 255) / 256), 256, 0, stream>>>(a, aBF, n8);
    k_agg<true, false><<<gridAG, 128, 0, stream>>>(aBF, xwT, b1, (void*)H);
    k_hw<<<gridMM, 128, 0, stream>>>(H, w2T, hwT);
    k_agg<true, true><<<gridAG, 128, 0, stream>>>(aBF, hwT, b2, d_out);
  } else {
    k_agg<false, false><<<gridAG, 128, 0, stream>>>(a, xwT, b1, (void*)H);
    k_hw<<<gridMM, 128, 0, stream>>>(H, w2T, hwT);
    k_agg<false, true><<<gridAG, 128, 0, stream>>>(a, hwT, b2, d_out);
  }
}
